// VanillaAttention_9947144257812
// MI455X (gfx1250) — matrix operand order repaired and checked by reading, not compiled, not run on hardware
//
#include <hip/hip_runtime.h>
#include <hip/hip_bf16.h>

// MI455X / gfx1250, wave32. bf16 WMMA (16x16x32, f32 accum) everywhere.
// Pipeline: 3x GEMM (proj) -> zero attn-mean -> fused scores/softmax/PV -> GEMM (Wo).
// Workspace usage: 40 MB (Qb 8MB bf16 + Kb 8MB bf16 + VTb 8MB bf16 + ctx 16MB f32).
// SCALE (1/sqrt(64) = 0.125) is folded into the Q projection (exact in bf16).

typedef __attribute__((ext_vector_type(16))) __bf16    v16bf;
typedef __attribute__((ext_vector_type(8)))  float     v8f;
typedef __attribute__((ext_vector_type(4)))  float     f4;
typedef __attribute__((ext_vector_type(4)))  unsigned  u4;
typedef __attribute__((ext_vector_type(4)))  int       v4i_;

#define B_  2
#define S_  2048
#define H_  1024
#define NH_ 16
#define HD_ 64

#if defined(__gfx1250__) && __has_builtin(__builtin_amdgcn_global_load_async_to_lds_b128) && __has_builtin(__builtin_amdgcn_s_wait_asynccnt)
#define HAVE_ASYNC_LDS 1
typedef __attribute__((address_space(1))) v4i_ g_v4i;   // global int4
typedef __attribute__((address_space(3))) v4i_ l_v4i;   // LDS int4
#else
#define HAVE_ASYNC_LDS 0
#endif

struct BfPair { u4 a; u4 b; };

static __device__ __forceinline__ v16bf frag2(const __bf16* p0, const __bf16* p1) {
  BfPair pr;
  pr.a = *(const u4*)p0;
  pr.b = *(const u4*)p1;
  return __builtin_bit_cast(v16bf, pr);
}
static __device__ __forceinline__ v16bf frag16(const __bf16* p) { return frag2(p, p + 8); }

static __device__ __forceinline__ v8f wmma_bf16(v16bf a, v16bf b, v8f c) {
  // D = A(16x32) * B(32x16) + C, f32 accumulate
  return __builtin_amdgcn_wmma_f32_16x16x32_bf16(false, a, false, b, (short)0, c, false, false);
}

union Pack16 { __bf16 h[16]; u4 u[2]; };

// ---------------------------------------------------------------------------
// GEMM: C[M=4096][N=1024] = A[4096][1024] @ W[1024][1024], fp32 in, bf16 WMMA.
// MODE 0: store bf16 at [b][h][s][d]                 (K)
// MODE 3: store bf16*0.125 at [b][h][s][d]           (Q, scale folded)
// MODE 1: store bf16 at [b][h][d][s], 16B packed     (V transposed)
// MODE 2: store fp32 row-major [m][n]                (final out)
// Block: 256 threads (8 waves). Tile 128(M) x 64(N), K-step 32.
// Double-buffered LDS: global loads of tile k+1 overlap WMMAs of tile k.
// ---------------------------------------------------------------------------
template <int MODE>
__global__ __launch_bounds__(256) void gemm_k(const float* __restrict__ A,
                                              const float* __restrict__ W,
                                              void* __restrict__ Dst) {
  __shared__ __bf16 a_t[2][128 * 32];   // [m][k]      16 KB
  __shared__ __bf16 b_t[2][64 * 32];    // [n][k] (T)   8 KB

  const int n0 = blockIdx.x * 64;
  const int m0 = blockIdx.y * 128;
  const int t  = threadIdx.x;
  const int w  = t >> 5;
  const int lane = t & 31;
  const int r  = lane & 15;
  const int hi = lane >> 4;

  // staging roles: A -> 16 floats/thread, W -> 8 floats/thread (transposed store)
  const int arow = t >> 1, ahalf = t & 1;
  const int bk   = t >> 3, bc    = (t & 7) * 8;
  const float* aptr = A + (size_t)(m0 + arow) * 1024 + ahalf * 16;
  const float* wptr = W + (size_t)bk * 1024 + n0 + bc;

  v8f acc[4];
#pragma unroll
  for (int i = 0; i < 4; ++i) acc[i] = v8f{0.f,0.f,0.f,0.f,0.f,0.f,0.f,0.f};

  f4 ar[4];
  f4 wr[2];

  auto stage_load = [&](int k0) {
#pragma unroll
    for (int j = 0; j < 4; ++j) ar[j] = *(const f4*)(aptr + k0 + j * 4);
#pragma unroll
    for (int j = 0; j < 2; ++j) wr[j] = *(const f4*)(wptr + (size_t)k0 * 1024 + j * 4);
  };
  auto stage_store = [&](int buf) {
    Pack16 p;
#pragma unroll
    for (int j = 0; j < 4; ++j) {
      p.h[j * 4 + 0] = (__bf16)ar[j].x;
      p.h[j * 4 + 1] = (__bf16)ar[j].y;
      p.h[j * 4 + 2] = (__bf16)ar[j].z;
      p.h[j * 4 + 3] = (__bf16)ar[j].w;
    }
    *(u4*)&a_t[buf][arow * 32 + ahalf * 16]     = p.u[0];
    *(u4*)&a_t[buf][arow * 32 + ahalf * 16 + 8] = p.u[1];
    b_t[buf][(bc + 0) * 32 + bk] = (__bf16)wr[0].x;
    b_t[buf][(bc + 1) * 32 + bk] = (__bf16)wr[0].y;
    b_t[buf][(bc + 2) * 32 + bk] = (__bf16)wr[0].z;
    b_t[buf][(bc + 3) * 32 + bk] = (__bf16)wr[0].w;
    b_t[buf][(bc + 4) * 32 + bk] = (__bf16)wr[1].x;
    b_t[buf][(bc + 5) * 32 + bk] = (__bf16)wr[1].y;
    b_t[buf][(bc + 6) * 32 + bk] = (__bf16)wr[1].z;
    b_t[buf][(bc + 7) * 32 + bk] = (__bf16)wr[1].w;
  };

  stage_load(0);
  stage_store(0);
  __syncthreads();

  for (int kt = 0; kt < 32; ++kt) {
    const int cur = kt & 1;
    if (kt + 1 < 32) {
      stage_load((kt + 1) * 32);          // in flight behind the WMMAs below
      if (kt + 2 < 32) __builtin_prefetch(aptr + (kt + 2) * 32, 0, 1);
    }
    {
      const __bf16* ap = &a_t[cur][(w * 16 + r) * 32];
      v16bf af = frag2(ap + hi * 16, ap + hi * 16 + 8);
#pragma unroll
      for (int nt = 0; nt < 4; ++nt) {
        v16bf bf = frag16(&b_t[cur][(nt * 16 + r) * 32 + hi * 16]);
        acc[nt] = wmma_bf16(af, bf, acc[nt]);
      }
    }
    if (kt + 1 < 32) stage_store(cur ^ 1);
    __syncthreads();
  }

  // --- epilogue: D tile layout: M = v + 8*hi (per VGPR), N = lane&15.
#pragma unroll
  for (int nt = 0; nt < 4; ++nt) {
    if (MODE == 1) {
      // [b][h][d][s]: the 8 accum rows are consecutive s -> one 16B store.
      const int n = n0 + nt * 16 + r;
      const int h = n >> 6, d = n & 63;
      const int m_base = m0 + w * 16 + 8 * hi;
      const int b = m_base >> 11, s = m_base & 2047;
      Pack16 p;
#pragma unroll
      for (int v = 0; v < 8; ++v) p.h[v] = (__bf16)acc[nt][v];
      *(u4*)&((__bf16*)Dst)[((((size_t)b * NH_ + h) * HD_ + d) * S_) + s] = p.u[0];
    } else {
#pragma unroll
      for (int v = 0; v < 8; ++v) {
        const int m = m0 + w * 16 + v + 8 * hi;
        const int n = n0 + nt * 16 + r;
        float val = acc[nt][v];
        if (MODE == 2) {
          ((float*)Dst)[(size_t)m * 1024 + n] = val;
        } else {
          if (MODE == 3) val *= 0.125f;   // fold attention scale into Q
          const int b = m >> 11, s = m & 2047;
          const int h = n >> 6,  d = n & 63;
          ((__bf16*)Dst)[((((size_t)b * NH_ + h) * S_ + s) * HD_) + d] = (__bf16)val;
        }
      }
    }
  }
}

// ---------------------------------------------------------------------------
// Zero-init the attn-mean region of d_out (harness poisons it with 0xAA).
// ---------------------------------------------------------------------------
__global__ __launch_bounds__(256) void zero_k(float* __restrict__ p, size_t n) {
  size_t i = ((size_t)blockIdx.x * 256 + threadIdx.x) * 4;
  if (i < n) *(f4*)(p + i) = f4{0.f, 0.f, 0.f, 0.f};
}

// ---------------------------------------------------------------------------
// Fused attention per (b, h, 32-query tile):
//   scores (32x2048 f32) resident in dynamic LDS (256KB of the 320KB WGP LDS)
//   -> softmax (wave32 shfl reductions) -> atomicAdd P/NH into attn-mean
//   -> ctx = P @ V via WMMA (V pre-transposed bf16).
// ---------------------------------------------------------------------------
__global__ __launch_bounds__(256) void attn_k(const __bf16* __restrict__ qb,
                                              const __bf16* __restrict__ kb,
                                              const __bf16* __restrict__ vtb,
                                              const int*    __restrict__ mask,
                                              float*        __restrict__ ctx,
                                              float*        __restrict__ attn_out) {
  extern __shared__ char smem[];
  float*  sc = (float*)smem;                       // [32][2048]
  __bf16* qs = (__bf16*)(smem + 32 * 2048 * 4);    // [32][64]

  const int qt = blockIdx.x, h = blockIdx.y, b = blockIdx.z;
  const int q0 = qt * 32;
  const int t = threadIdx.x, w = t >> 5, lane = t & 31;
  const int r = lane & 15, hi = lane >> 4;

  // Load Q tile (32x64 bf16, pre-scaled by 0.125) into LDS: 16B per thread.
  const __bf16* qbase = qb + ((((size_t)b * NH_ + h) * S_) + q0) * HD_;
#if HAVE_ASYNC_LDS
  __builtin_amdgcn_global_load_async_to_lds_b128(
      (g_v4i*)(void*)&qbase[t * 8],
      (l_v4i*)(void*)&qs[t * 8], 0, 0);
  __builtin_amdgcn_s_wait_asynccnt(0);
#else
  *(u4*)&qs[t * 8] = *(const u4*)&qbase[t * 8];
#endif

  // Preload this lane's 16 mask words (one clause; keeps score loop pure WMMA).
  int mvx[16];
#pragma unroll
  for (int nt = 0; nt < 16; ++nt) mvx[nt] = mask[b * S_ + w * 256 + nt * 16 + r];

  __syncthreads();

  // ---- Phase 1: scores. Wave w owns keys [w*256, w*256+256).
  const __bf16* kbase = kb + (((size_t)b * NH_ + h) * S_) * HD_;
#pragma unroll
  for (int mt = 0; mt < 2; ++mt) {
    const __bf16* qp = &qs[(mt * 16 + r) * 64];
    v16bf a0 = frag2(qp + hi * 16,      qp + hi * 16 + 8);       // d 0..31
    v16bf a1 = frag2(qp + 32 + hi * 16, qp + 40 + hi * 16);      // d 32..63
#pragma unroll
    for (int nt = 0; nt < 16; ++nt) {
      const int key = w * 256 + nt * 16 + r;
      const __bf16* kp = kbase + (size_t)key * HD_;
      v16bf b0 = frag16(kp + hi * 16);
      v16bf b1 = frag16(kp + 32 + hi * 16);
      v8f s = v8f{0.f,0.f,0.f,0.f,0.f,0.f,0.f,0.f};
      s = wmma_bf16(a0, b0, s);
      s = wmma_bf16(a1, b1, s);
#pragma unroll
      for (int v = 0; v < 8; ++v) {
        float sv = s[v];                 // scale already folded into Q
        if (mvx[nt] == 0) sv = -1.0e9f;
        sc[(mt * 16 + v + 8 * hi) * 2048 + key] = sv;
      }
    }
  }
  __syncthreads();

  // ---- Phase 2: softmax. Wave w owns rows [4w, 4w+4).
  for (int rr = 0; rr < 4; ++rr) {
    const int row = w * 4 + rr;
    float* pr = sc + row * 2048;
    float mx = -1.0e30f;
    for (int i = lane; i < 2048; i += 32) mx = fmaxf(mx, pr[i]);
#pragma unroll
    for (int o = 16; o > 0; o >>= 1) mx = fmaxf(mx, __shfl_xor(mx, o, 32));
    float sum = 0.f;
    for (int i = lane; i < 2048; i += 32) {
      float p = __expf(pr[i] - mx);
      pr[i] = p;
      sum += p;
    }
#pragma unroll
    for (int o = 16; o > 0; o >>= 1) sum += __shfl_xor(sum, o, 32);
    const float inv = 1.0f / sum;
    float* arow = attn_out + ((size_t)(b * S_) + q0 + row) * (size_t)S_;
    for (int i = lane; i < 2048; i += 32) {
      float p = pr[i] * inv;
      pr[i] = p;                                   // normalized P for P@V
      atomicAdd(&arow[i], p * (1.0f / NH_));       // head-mean accumulation
    }
  }
  __syncthreads();

  // ---- Phase 3: ctx = P @ V. Wave w -> (mt = w>>2, dt = w&3) 16x16 ctx tile.
  const int mt = w >> 2, dt = w & 3;
  const __bf16* vbase = vtb + (((size_t)b * NH_ + h) * HD_ + dt * 16) * S_;
  v8f c = v8f{0.f,0.f,0.f,0.f,0.f,0.f,0.f,0.f};
  for (int kc = 0; kc < 64; ++kc) {
    const float* pr = sc + (mt * 16 + r) * 2048 + kc * 32;
    f4 c0 = *(const f4*)(pr + hi * 16);
    f4 c1 = *(const f4*)(pr + hi * 16 + 4);
    f4 c2 = *(const f4*)(pr + hi * 16 + 8);
    f4 c3 = *(const f4*)(pr + hi * 16 + 12);
    v16bf a;
    a[0]=(__bf16)c0.x; a[1]=(__bf16)c0.y; a[2]=(__bf16)c0.z; a[3]=(__bf16)c0.w;
    a[4]=(__bf16)c1.x; a[5]=(__bf16)c1.y; a[6]=(__bf16)c1.z; a[7]=(__bf16)c1.w;
    a[8]=(__bf16)c2.x; a[9]=(__bf16)c2.y; a[10]=(__bf16)c2.z; a[11]=(__bf16)c2.w;
    a[12]=(__bf16)c3.x; a[13]=(__bf16)c3.y; a[14]=(__bf16)c3.z; a[15]=(__bf16)c3.w;
    v16bf bfr = frag16(vbase + (size_t)r * S_ + kc * 32 + hi * 16);
    c = wmma_bf16(a, bfr, c);
  }
  const int d = dt * 16 + r;
#pragma unroll
  for (int v = 0; v < 8; ++v) {
    const int q = q0 + mt * 16 + v + 8 * hi;
    ctx[((size_t)b * S_ + q) * (size_t)H_ + h * HD_ + d] = c[v];
  }
}

// ---------------------------------------------------------------------------
extern "C" void kernel_launch(void* const* d_in, const int* in_sizes, int n_in,
                              void* d_out, int out_size, void* d_ws, size_t ws_size,
                              hipStream_t stream) {
  const float* Xq = (const float*)d_in[0];
  const float* Xk = (const float*)d_in[1];
  const float* Xv = (const float*)d_in[2];
  const int*   mk = (const int*)  d_in[3];
  const float* Wq = (const float*)d_in[4];
  const float* Wk = (const float*)d_in[5];
  const float* Wv = (const float*)d_in[6];
  const float* Wo = (const float*)d_in[7];

  float* out      = (float*)d_out;
  float* attn_out = out + (size_t)B_ * S_ * H_;          // (B,S,S) head-mean

  char* ws = (char*)d_ws;                                 // needs >= 40 MB
  __bf16* qb  = (__bf16*)(ws);
  __bf16* kb  = (__bf16*)(ws + ((size_t)8  << 20));
  __bf16* vtb = (__bf16*)(ws + ((size_t)16 << 20));
  float*  ctx = (float*) (ws + ((size_t)24 << 20));

  const dim3 gg(16, 32), bb(256);
  gemm_k<3><<<gg, bb, 0, stream>>>(Xq, Wq, (void*)qb);   // Q (scale folded)
  gemm_k<0><<<gg, bb, 0, stream>>>(Xk, Wk, (void*)kb);   // K
  gemm_k<1><<<gg, bb, 0, stream>>>(Xv, Wv, (void*)vtb);  // V transposed

  zero_k<<<8192, 256, 0, stream>>>(attn_out, (size_t)B_ * S_ * S_);

  const size_t smem = (size_t)32 * 2048 * 4 + (size_t)32 * 64 * 2; // 260 KB
  attn_k<<<dim3(S_ / 32, NH_, B_), 256, smem, stream>>>(qb, kb, vtb, mk, ctx, attn_out);

  gemm_k<2><<<gg, bb, 0, stream>>>(ctx, Wo, (void*)out);
}
